// SwinTransformerBlock_75084618268795
// MI455X (gfx1250) — matrix operand order repaired and checked by reading, not compiled, not run on hardware
//
#include <hip/hip_runtime.h>
#include <hip/hip_fp16.h>

// ---------------------------------------------------------------------------
// CDNA5 (gfx1250) Swin block: wave32 + v_wmma_f32_16x16x32_f16 everywhere.
// ---------------------------------------------------------------------------

typedef __attribute__((ext_vector_type(16))) _Float16 v16h;
typedef __attribute__((ext_vector_type(8)))  _Float16 v8h;
typedef __attribute__((ext_vector_type(8)))  float    v8f;

#define WMMA16(a, b, c) \
  __builtin_amdgcn_wmma_f32_16x16x32_f16(false, (a), false, (b), (short)0, (c), false, false)

// ----- problem constants ----------------------------------------------------
constexpr int kB     = 8;
constexpr int kH     = 64;
constexpr int kW     = 64;
constexpr int kC     = 512;
constexpr int kHeads = 16;
constexpr int kHD    = 32;
constexpr int kNtok  = 256;                 // tokens per window
constexpr int kNW    = 16;                  // windows per image
constexpr int kBW    = kB * kNW;            // 128 window instances
constexpr int kTok   = kB * kH * kW;        // 32768 tokens
constexpr float kScale = 0.17677669529663687f;  // 32^-0.5

// ----- WMMA fragment loaders (contiguous k = 16*(lane/16) + s in A and B) ---------------------------
// A: 16x32 f16, row-major source, lda in elements.
__device__ __forceinline__ v16h load_a16(const _Float16* A, int lda, int m0, int k0, int lane) {
  const int lo = (lane & 16) ? 16 : 0;                // K 0..15 / 16..31 halves
  const _Float16* p = A + (size_t)(m0 + (lane & 15)) * lda + k0;
  union { v16h v; v8h h[2]; } u;
  u.h[0] = *(const v8h*)(p + lo);                     // K lo .. lo+7
  u.h[1] = *(const v8h*)(p + lo + 8);                 // K lo+8 .. lo+15
  return u.v;
}
// B: 32x16 (KxN) fragment from an N-major ("transposed") array: Bt[n*ldb + k].
__device__ __forceinline__ v16h load_b16(const _Float16* Bt, int ldb, int n0, int k0, int lane) {
  const int ko = (lane & 16) ? 16 : 0;                // lanes 0-15: K 0-15, 16-31: K 16-31
  const _Float16* p = Bt + (size_t)(n0 + (lane & 15)) * ldb + k0 + ko;
  return *(const v16h*)p;
}

// ----- weight convert + transpose: out[n*K + k] = (f16) in[k*N + n] ---------
__global__ void cvt_transpose_kernel(const float* __restrict__ in, _Float16* __restrict__ out,
                                     int K, int N) {
  int idx = blockIdx.x * 256 + threadIdx.x;
  if (idx >= K * N) return;
  int n = idx / K, k = idx - n * K;
  out[idx] = (_Float16)in[(size_t)k * N + n];
}

// ----- LayerNorm (+ optional shift-roll + window partition) -----------------
template <bool PARTITION>
__global__ __launch_bounds__(128) void ln_kernel(const float* __restrict__ x,
                                                 const float* __restrict__ g,
                                                 const float* __restrict__ b,
                                                 _Float16* __restrict__ out) {
  __shared__ float s1[128], s2[128];
  const int t = threadIdx.x;
  const size_t tok = blockIdx.x;
  const float4 v = ((const float4*)(x + tok * kC))[t];
  s1[t] = v.x + v.y + v.z + v.w;
  s2[t] = v.x * v.x + v.y * v.y + v.z * v.z + v.w * v.w;
  __syncthreads();
  for (int off = 64; off > 0; off >>= 1) {
    if (t < off) { s1[t] += s1[t + off]; s2[t] += s2[t + off]; }
    __syncthreads();
  }
  const float mu   = s1[0] * (1.0f / kC);
  const float var  = s2[0] * (1.0f / kC) - mu * mu;
  const float rstd = rsqrtf(var + 1e-5f);

  size_t orow = tok;
  if (PARTITION) {
    const int bi = (int)(tok >> 12);
    const int rem = (int)(tok & 4095);
    const int hh = rem >> 6, ww = rem & 63;
    const int i = (hh + 56) & 63, j = (ww + 56) & 63;   // roll(-8,-8)
    const int wrow = (i >> 4) * 4 + (j >> 4);
    const int n = (i & 15) * 16 + (j & 15);
    orow = ((size_t)bi * kNW + wrow) * kNtok + n;
  }
  const int c = t * 4;
  _Float16* o = out + orow * kC + c;
  o[0] = (_Float16)((v.x - mu) * rstd * g[c + 0] + b[c + 0]);
  o[1] = (_Float16)((v.y - mu) * rstd * g[c + 1] + b[c + 1]);
  o[2] = (_Float16)((v.z - mu) * rstd * g[c + 2] + b[c + 2]);
  o[3] = (_Float16)((v.w - mu) * rstd * g[c + 3] + b[c + 3]);
}

// ----- generic WMMA GEMM: C = A(f16, MxK) * Bt(f16, NxK)^T + bias -----------
// 8 waves/block, each wave owns a 32x32 tile -> block tile 128(M) x 64(N).
// EPI: 0 = QKV scatter, 1 = proj + window-reverse + residual,
//      2 = exact GELU -> f16, 3 = + residual -> f32 out.
template <int EPI>
__global__ __launch_bounds__(256) void gemm_wmma_kernel(
    const _Float16* __restrict__ A, const _Float16* __restrict__ Bt,
    const float* __restrict__ bias, int M, int N, int K,
    void* e0, void* e1, const void* e2) {
  const int lane = threadIdx.x & 31;
  const int wv   = threadIdx.x >> 5;
  const int m0   = blockIdx.x * 128 + (wv & 3) * 32;
  const int n0   = blockIdx.y * 64 + (wv >> 2) * 32;
  const int hl   = lane >> 4, lc = lane & 15;

  v8f acc[2][2] = {};
  for (int k0 = 0; k0 < K; k0 += 32) {
    const v16h a0 = load_a16(A, K, m0,      k0, lane);
    const v16h a1 = load_a16(A, K, m0 + 16, k0, lane);
    const v16h b0 = load_b16(Bt, K, n0,      k0, lane);
    const v16h b1 = load_b16(Bt, K, n0 + 16, k0, lane);
    acc[0][0] = WMMA16(a0, b0, acc[0][0]);
    acc[0][1] = WMMA16(a0, b1, acc[0][1]);
    acc[1][0] = WMMA16(a1, b0, acc[1][0]);
    acc[1][1] = WMMA16(a1, b1, acc[1][1]);
  }

#pragma unroll
  for (int mi = 0; mi < 2; ++mi)
#pragma unroll
    for (int ni = 0; ni < 2; ++ni)
#pragma unroll
      for (int v = 0; v < 8; ++v) {
        const int row = m0 + mi * 16 + v + 8 * hl;
        const int col = n0 + ni * 16 + lc;
        const float val = acc[mi][ni][v] + bias[col];

        if constexpr (EPI == 0) {           // QKV scatter into Q/K/Vt head layout
          _Float16* Q  = (_Float16*)e0;
          _Float16* Km = (_Float16*)e1;
          _Float16* Vt = (_Float16*)const_cast<void*>(e2);
          const int bw = row >> 8, n = row & 255;
          const int s = col >> 9, hd = col & 511;
          const int head = hd >> 5, d = hd & 31;
          const size_t bh = (size_t)bw * kHeads + head;
          if (s == 0)      Q [(bh * kNtok + n) * kHD + d] = (_Float16)(val * kScale);
          else if (s == 1) Km[(bh * kNtok + n) * kHD + d] = (_Float16)val;
          else             Vt[(bh * kHD + d) * kNtok + n] = (_Float16)val;
        } else if constexpr (EPI == 1) {    // proj: window reverse + roll(+8) + residual
          float* xmid = (float*)e0;
          const float* xin = (const float*)e2;
          const int bw = row >> 8, bi = bw >> 4, w = bw & 15, n = row & 255;
          const int i = (w >> 2) * 16 + (n >> 4);
          const int j = (w & 3) * 16 + (n & 15);
          const int hh = (i + 8) & 63, ww = (j + 8) & 63;
          const size_t tok = ((size_t)bi << 12) + hh * 64 + ww;
          xmid[tok * kC + col] = xin[tok * kC + col] + val;
        } else if constexpr (EPI == 2) {    // FC1 + exact GELU
          _Float16* Hh = (_Float16*)e0;
          const float gl = 0.5f * val * (1.0f + erff(val * 0.7071067811865475f));
          Hh[(size_t)row * N + col] = (_Float16)gl;
        } else {                            // FC2 + residual
          float* outp = (float*)e0;
          const float* xmid = (const float*)e2;
          outp[(size_t)row * kC + col] = xmid[(size_t)row * kC + col] + val;
        }
      }
}

// ----- attention: one (window, head) per block; 16 waves x one 16-row strip -
__device__ __forceinline__ int rgn3(int a) { return (a >= 56) ? 2 : ((a >= 48) ? 1 : 0); }

__global__ __launch_bounds__(512) void attn_kernel(
    const _Float16* __restrict__ Q, const _Float16* __restrict__ Km,
    const _Float16* __restrict__ Vt, const float* __restrict__ rpb,
    _Float16* __restrict__ O) {
  extern __shared__ _Float16 Pbuf[];              // 256 x 256 f16 = 128 KB
  const int lane = threadIdx.x & 31;
  const int wm   = threadIdx.x >> 5;              // M-tile 0..15
  const int bwh  = blockIdx.x;                    // 0..2047
  const int bw   = bwh >> 4, h = bwh & 15;
  const size_t base = (size_t)bwh * kNtok * kHD;
  const int hl = lane >> 4, lc = lane & 15;
  const int w  = bw & 15, wi = w >> 2, wj = w & 3;

  // S = (q * scale) @ k^T : K = 32 = head dim -> one WMMA per 16x16 tile.
  const v16h aq = load_a16(Q + base, kHD, wm * 16, 0, lane);
  v8f s[16];
#pragma unroll
  for (int t = 0; t < 16; ++t) {
    const v16h bk = load_b16(Km + base, kHD, t * 16, 0, lane);
    v8f z = {};
    s[t] = WMMA16(aq, bk, z);
  }

  // relative-position bias + shift mask + row max (rows live in half-waves)
  float mx[8], sm[8];
#pragma unroll
  for (int v = 0; v < 8; ++v) {
    const int n = wm * 16 + v + 8 * hl;
    const int nr = n >> 4, nc = n & 15;
    const int regn = rgn3(wi * 16 + nr) * 3 + rgn3(wj * 16 + nc);
    float m_ = -3.0e38f;
#pragma unroll
    for (int t = 0; t < 16; ++t) {
      const int m = t * 16 + lc;
      const int mr = m >> 4, mc = m & 15;
      const int regm = rgn3(wi * 16 + mr) * 3 + rgn3(wj * 16 + mc);
      const int rel = (nr - mr + 15) * 31 + (nc - mc + 15);
      float sv = s[t][v] + rpb[rel * kHeads + h];
      if (regn != regm) sv -= 100.0f;
      s[t][v] = sv;
      m_ = fmaxf(m_, sv);
    }
#pragma unroll
    for (int d = 8; d >= 1; d >>= 1) m_ = fmaxf(m_, __shfl_xor(m_, d, 32));
    mx[v] = m_;
  }
  // exp + row sum
#pragma unroll
  for (int v = 0; v < 8; ++v) {
    float a_ = 0.0f;
#pragma unroll
    for (int t = 0; t < 16; ++t) {
      const float e = __expf(s[t][v] - mx[v]);
      s[t][v] = e;
      a_ += e;
    }
#pragma unroll
    for (int d = 8; d >= 1; d >>= 1) a_ += __shfl_xor(a_, d, 32);
    sm[v] = 1.0f / a_;
  }
  // normalized P -> LDS (f16, row-major 256x256)
#pragma unroll
  for (int v = 0; v < 8; ++v) {
    const int n = wm * 16 + v + 8 * hl;
#pragma unroll
    for (int t = 0; t < 16; ++t)
      Pbuf[(size_t)n * kNtok + t * 16 + lc] = (_Float16)(s[t][v] * sm[v]);
  }
  __syncthreads();

  // O = P @ V : 16x32 strip per wave, 8 chained k-steps per 16x16 tile.
  const _Float16* Prow = Pbuf + (size_t)wm * 16 * kNtok;
#pragma unroll
  for (int ni = 0; ni < 2; ++ni) {
    v8f o = {};
#pragma unroll
    for (int ks = 0; ks < 8; ++ks) {
      const v16h ap = load_a16(Prow, kNtok, 0, ks * 32, lane);
      const v16h bv = load_b16(Vt + base, kNtok, ni * 16, ks * 32, lane);
      o = WMMA16(ap, bv, o);
    }
#pragma unroll
    for (int v = 0; v < 8; ++v) {
      const int n = wm * 16 + v + 8 * hl;
      const size_t R = (size_t)bw * kNtok + n;
      O[R * kC + h * kHD + ni * 16 + lc] = (_Float16)o[v];
    }
  }
}

// ---------------------------------------------------------------------------
extern "C" void kernel_launch(void* const* d_in, const int* in_sizes, int n_in,
                              void* d_out, int out_size, void* d_ws, size_t ws_size,
                              hipStream_t stream) {
  (void)in_sizes; (void)n_in; (void)out_size; (void)ws_size;
  const float* x     = (const float*)d_in[0];
  const float* rpb   = (const float*)d_in[1];
  const float* wqkv  = (const float*)d_in[2];
  const float* bqkv  = (const float*)d_in[3];
  const float* wproj = (const float*)d_in[4];
  const float* bproj = (const float*)d_in[5];
  const float* n1g   = (const float*)d_in[6];
  const float* n1b   = (const float*)d_in[7];
  const float* n2g   = (const float*)d_in[8];
  const float* n2b   = (const float*)d_in[9];
  const float* wfc1  = (const float*)d_in[10];
  const float* bfc1  = (const float*)d_in[11];
  const float* wfc2  = (const float*)d_in[12];
  const float* bfc2  = (const float*)d_in[13];
  float* out = (float*)d_out;

  // workspace layout (bytes); Hh aliases Kh+Vt tail (both dead after attention)
  char* ws = (char*)d_ws;
  _Float16* Wqkvt = (_Float16*)(ws + 0);            // 1536x512  f16
  _Float16* Wprjt = (_Float16*)(ws + 1572864);      //  512x512  f16
  _Float16* Wf1t  = (_Float16*)(ws + 2097152);      // 2048x512  f16
  _Float16* Wf2t  = (_Float16*)(ws + 4194304);      //  512x2048 f16
  _Float16* Ah    = (_Float16*)(ws + 6291456);      // 32768x512 f16 (later Oh)
  _Float16* Qh    = (_Float16*)(ws + 39845888);     // 2048x256x32 f16 (later A2h)
  float*    Xmid  = (float*)   (ws + 73400320);     // 32768x512 f32
  _Float16* Kh    = (_Float16*)(ws + 140509184);    // 2048x256x32 f16
  _Float16* Vt    = (_Float16*)(ws + 174063616);    // 2048x32x256 f16
  _Float16* Oh    = Ah;                             // attention out (Ah is dead)
  _Float16* A2h   = Qh;                             // LN2 out (Qh is dead)
  _Float16* Hh    = Kh;                             // 32768x2048 f16, aliases Kh+Vt+tail

  const int M = kTok;

  // 1) weight convert + N-major transpose (once per launch)
  cvt_transpose_kernel<<<(512 * 1536 + 255) / 256, 256, 0, stream>>>(wqkv, Wqkvt, 512, 1536);
  cvt_transpose_kernel<<<(512 *  512 + 255) / 256, 256, 0, stream>>>(wproj, Wprjt, 512, 512);
  cvt_transpose_kernel<<<(512 * 2048 + 255) / 256, 256, 0, stream>>>(wfc1, Wf1t, 512, 2048);
  cvt_transpose_kernel<<<(2048 * 512 + 255) / 256, 256, 0, stream>>>(wfc2, Wf2t, 2048, 512);

  // 2) LN1 + roll(-8,-8) + window partition -> f16
  ln_kernel<true><<<kTok, 128, 0, stream>>>(x, n1g, n1b, Ah);

  // 3) QKV GEMM (32768x1536x512) with head-layout scatter epilogue
  gemm_wmma_kernel<0><<<dim3(M / 128, 1536 / 64), 256, 0, stream>>>(
      Ah, Wqkvt, bqkv, M, 1536, 512, Qh, Kh, Vt);

  // 4) attention: 2048 (window, head) blocks, 128 KB LDS for P
  attn_kernel<<<kBW * kHeads, 512, 256 * 256 * sizeof(_Float16), stream>>>(Qh, Kh, Vt, rpb, Oh);

  // 5) proj GEMM + window reverse + roll(+8,+8) + residual -> Xmid
  gemm_wmma_kernel<1><<<dim3(M / 128, 512 / 64), 256, 0, stream>>>(
      Oh, Wprjt, bproj, M, 512, 512, Xmid, nullptr, x);

  // 6) LN2 -> f16
  ln_kernel<false><<<kTok, 128, 0, stream>>>(Xmid, n2g, n2b, A2h);

  // 7) FC1 GEMM + exact GELU -> f16
  gemm_wmma_kernel<2><<<dim3(M / 128, 2048 / 64), 256, 0, stream>>>(
      A2h, Wf1t, bfc1, M, 2048, 512, Hh, nullptr, nullptr);

  // 8) FC2 GEMM + residual -> f32 out
  gemm_wmma_kernel<3><<<dim3(M / 128, 512 / 64), 256, 0, stream>>>(
      Hh, Wf2t, bfc2, M, 512, 2048, out, nullptr, Xmid);
}
